// FFM_20830591385963
// MI455X (gfx1250) — compile-verified, matrix operand order checked by reading, not run on hardware
//
#include <hip/hip_runtime.h>
#include <hip/hip_bf16.h>

// Problem constants (from reference)
#define BB 4096
#define NN 2048
#define FF 64
#define KK 64
#define NWAVES (BB / 16)   // 256 waves in the main pass

typedef __attribute__((ext_vector_type(16))) __bf16 v16bf;
typedef __attribute__((ext_vector_type(8)))  float  v8f;

// ---------------------------------------------------------------------------
// Math: out[b] = X[b,:].W + bias + 0.5*( ||X V||_F^2 - sum_i ||V_i||^2 ||X[:,i]||^2 )
// with V[i] = vector[i, f2f[i]].  Single fused pass over X (32 MB, the only
// large operand) -> bandwidth-bound at ~1.4us on 23.3 TB/s HBM.
//
// Workspace layout (bytes):
//   [0)        Vb        : bf16 [N*K] pre-packed WMMA-B fragments  (262144 B)
//   [262144)   vnorm2    : f32  [N]                                (  8192 B)
//   [270336)   lin       : f32  [B]                                ( 16384 B)
//   [286720)   frob_part : f32  [NWAVES]                           (  1024 B)
//   [287744)   diag_part : f32  [NWAVES]                           (  1024 B)
// ---------------------------------------------------------------------------

__device__ __forceinline__ float dot4(float4 a, float4 b) {
    return a.x * b.x + a.y * b.y + a.z * b.z + a.w * b.w;
}
__device__ __forceinline__ float wsq4(float4 v, float4 x) {
    return v.x * x.x * x.x + v.y * x.y * x.y + v.z * x.z * x.z + v.w * x.w * x.w;
}

// ---------------------------------------------------------------------------
// Kernel 1: gather V[i] = vector[i, f2f[i]], compute ||V_i||^2, and pack V
// into bf16 WMMA-B fragments matching the ISA 16-bit fragment striping
// (lanes 0-15 hold K={0..7,16..23}, lanes 16-31 hold K={8..15,24..31}).
// ---------------------------------------------------------------------------
__global__ __launch_bounds__(256) void ffm_gather_pack(
    const float* __restrict__ vector, const int* __restrict__ f2f,
    __bf16* __restrict__ Vb, float* __restrict__ vnorm2) {
    const int n = blockIdx.x * blockDim.x + threadIdx.x;
    if (n >= NN) return;

    const int f = f2f[n];
    const float* src = vector + ((size_t)n * FF + f) * KK;

    const int c = n >> 5;                      // contraction chunk
    const int r = n & 31;                      // row within chunk
    const int lane_off = (r & 8) ? 16 : 0;
    const int elem = (r & 7) + ((r & 16) ? 8 : 0);

    float acc = 0.0f;
#pragma unroll
    for (int k = 0; k < KK; ++k) {
        const float v = src[k];
        acc += v * v;
        const int t = k >> 4;                  // k-tile
        const int j = k & 15;                  // column within tile
        const int lane = lane_off + j;
        Vb[(((size_t)(c * 4 + t) * 32) + lane) * 16 + elem] = (__bf16)v;
    }
    vnorm2[n] = acc;
}

// ---------------------------------------------------------------------------
// Kernel 2: single fused pass over X.  One wave32 per 16-row batch tile;
// loops N in chunks of 32: 4x global_load_b128 of f32 X -> bf16 A fragment,
// 4x v_wmma_f32_16x16x32_bf16 against resident V fragments, plus f32
// linear (X.W) and diag (vnorm2 * X^2) partials.  Deterministic epilogue:
// per-wave partials written to workspace (no float atomics).
// ---------------------------------------------------------------------------
__global__ __launch_bounds__(256) void ffm_main(
    const float* __restrict__ X, const float* __restrict__ W,
    const __bf16* __restrict__ Vb, const float* __restrict__ vnorm2,
    float* __restrict__ lin, float* __restrict__ frob_part,
    float* __restrict__ diag_part) {
    const int lane = threadIdx.x & 31;
    const int wave = threadIdx.x >> 5;
    const int gw   = blockIdx.x * 8 + wave;      // 0..255
    const int row0 = gw * 16;
    const int m    = lane & 15;                  // batch row within tile
    const int hi   = lane >> 4;                  // lane group
    const int boff = hi * 8;                     // column sub-offset

    const float* xrow = X + (size_t)(row0 + m) * NN;

    v8f c0 = {}, c1 = {}, c2 = {}, c3 = {};
    float linp = 0.0f, diagp = 0.0f;

    const v16bf* fragbase = (const v16bf*)Vb;

    for (int n0 = 0; n0 < NN; n0 += 32) {
        const float4* p0 = (const float4*)(xrow + n0 + boff);
        const float4* p1 = (const float4*)(xrow + n0 + 16 + boff);
        float4 x0 = p0[0], x1 = p0[1];
        float4 x2 = p1[0], x3 = p1[1];

        if (n0 + 32 < NN)  // global_prefetch_b8 on the next X chunk
            __builtin_prefetch(xrow + n0 + 32 + boff, 0, 0);

        // W / vnorm2 chunks (broadcast across lane groups; WGP$-resident)
        const float4* wp0 = (const float4*)(W + n0 + boff);
        const float4* wp1 = (const float4*)(W + n0 + 16 + boff);
        float4 w0 = wp0[0], w1 = wp0[1], w2 = wp1[0], w3 = wp1[1];
        const float4* vp0 = (const float4*)(vnorm2 + n0 + boff);
        const float4* vp1 = (const float4*)(vnorm2 + n0 + 16 + boff);
        float4 q0 = vp0[0], q1 = vp0[1], q2 = vp1[0], q3 = vp1[1];

        linp  += dot4(x0, w0) + dot4(x1, w1) + dot4(x2, w2) + dot4(x3, w3);
        diagp += wsq4(q0, x0) + wsq4(q1, x1) + wsq4(q2, x2) + wsq4(q3, x3);

        // A fragment (bf16): elems 0..7 = first col group, 8..15 = second
        v16bf a;
        a[0]  = (__bf16)x0.x; a[1]  = (__bf16)x0.y; a[2]  = (__bf16)x0.z; a[3]  = (__bf16)x0.w;
        a[4]  = (__bf16)x1.x; a[5]  = (__bf16)x1.y; a[6]  = (__bf16)x1.z; a[7]  = (__bf16)x1.w;
        a[8]  = (__bf16)x2.x; a[9]  = (__bf16)x2.y; a[10] = (__bf16)x2.z; a[11] = (__bf16)x2.w;
        a[12] = (__bf16)x3.x; a[13] = (__bf16)x3.y; a[14] = (__bf16)x3.z; a[15] = (__bf16)x3.w;

        // 4 pre-packed B fragments for this contraction chunk (32B/lane each)
        const v16bf* fb = fragbase + (size_t)(n0 >> 5) * 128;
        v16bf b0 = fb[lane];
        v16bf b1 = fb[32 + lane];
        v16bf b2 = fb[64 + lane];
        v16bf b3 = fb[96 + lane];

        c0 = __builtin_amdgcn_wmma_f32_16x16x32_bf16(false, a, false, b0, (short)0, c0, false, false);
        c1 = __builtin_amdgcn_wmma_f32_16x16x32_bf16(false, a, false, b1, (short)0, c1, false, false);
        c2 = __builtin_amdgcn_wmma_f32_16x16x32_bf16(false, a, false, b2, (short)0, c2, false, false);
        c3 = __builtin_amdgcn_wmma_f32_16x16x32_bf16(false, a, false, b3, (short)0, c3, false, false);
    }

    // Frobenius-norm partial of this wave's 16x64 tile of M = X*V
    float fp = 0.0f;
#pragma unroll
    for (int i = 0; i < 8; ++i)
        fp += c0[i] * c0[i] + c1[i] * c1[i] + c2[i] * c2[i] + c3[i] * c3[i];

    // deterministic wave32 reductions (fixed xor-tree order)
#pragma unroll
    for (int off = 16; off >= 1; off >>= 1) {
        fp    += __shfl_xor(fp, off, 32);
        diagp += __shfl_xor(diagp, off, 32);
    }
    if (lane == 0) {
        frob_part[gw] = fp;
        diag_part[gw] = diagp;
    }

    // linear term: row m = lane m partial + lane m+16 partial
    float l2 = linp + __shfl_down(linp, 16, 32);
    if (hi == 0) lin[row0 + m] = l2;
}

// ---------------------------------------------------------------------------
// Kernel 3: deterministically reduce the 256 per-wave partials inside each
// 256-thread block (shfl tree + LDS), then out[b] = lin[b] + bias + inter.
// ---------------------------------------------------------------------------
__global__ __launch_bounds__(256) void ffm_final(
    const float* __restrict__ lin, const float* __restrict__ bias,
    const float* __restrict__ frob_part, const float* __restrict__ diag_part,
    float* __restrict__ out) {
    __shared__ float sfr[8], sdg[8];
    const int lane = threadIdx.x & 31;
    const int wave = threadIdx.x >> 5;

    float f = frob_part[threadIdx.x];
    float d = diag_part[threadIdx.x];
#pragma unroll
    for (int off = 16; off >= 1; off >>= 1) {
        f += __shfl_xor(f, off, 32);
        d += __shfl_xor(d, off, 32);
    }
    if (lane == 0) { sfr[wave] = f; sdg[wave] = d; }
    __syncthreads();
    if (threadIdx.x == 0) {
        float tf = 0.0f, td = 0.0f;
#pragma unroll
        for (int i = 0; i < 8; ++i) { tf += sfr[i]; td += sdg[i]; }
        sfr[0] = tf; sdg[0] = td;
    }
    __syncthreads();

    const float inter = 0.5f * (sfr[0] - sdg[0]);
    const int i = blockIdx.x * blockDim.x + threadIdx.x;
    if (i < BB) out[i] = lin[i] + bias[0] + inter;
}

extern "C" void kernel_launch(void* const* d_in, const int* in_sizes, int n_in,
                              void* d_out, int out_size, void* d_ws, size_t ws_size,
                              hipStream_t stream) {
    const float* X      = (const float*)d_in[0];  // [B, N]
    const float* W      = (const float*)d_in[1];  // [1, N]
    const float* bias   = (const float*)d_in[2];  // [1]
    const float* vector = (const float*)d_in[3];  // [N, F, K]
    const int*   f2f    = (const int*)d_in[4];    // [N]
    float* out = (float*)d_out;                   // [B]

    char* ws = (char*)d_ws;
    __bf16* Vb        = (__bf16*)(ws);                // 262144 B
    float*  vnorm2    = (float*)(ws + 262144);        //   8192 B
    float*  lin       = (float*)(ws + 270336);        //  16384 B
    float*  frob_part = (float*)(ws + 286720);        //   1024 B
    float*  diag_part = (float*)(ws + 287744);        //   1024 B

    ffm_gather_pack<<<NN / 256, 256, 0, stream>>>(vector, f2f, Vb, vnorm2);
    ffm_main<<<NWAVES / 8, 256, 0, stream>>>(X, W, Vb, vnorm2, lin, frob_part, diag_part);
    ffm_final<<<BB / 256, 256, 0, stream>>>(lin, bias, frob_part, diag_part, out);
}
